// MPNN_50680614093673
// MI455X (gfx1250) — matrix operand order repaired and checked by reading, not compiled, not run on hardware
//
#include <hip/hip_runtime.h>

typedef __attribute__((ext_vector_type(16))) __bf16 v16bf;
typedef __attribute__((ext_vector_type(8)))  float  v8f;

#define N_NODES 8000
#define NEDGE   64000
#define DIN     16
#define H       128
#define MDIM    16
#define EDF     4
#define DOUT    4
#define NLAYERS 2
#define BN_EPS  1e-5f

#define ETILE 64          // edges per block (4 WMMA M-tiles)
#define HSTR  (ETILE + 4) // padded LDS stride for gathered h (bank spread, 16B align)

// ---------------------------------------------------------------- utilities
__global__ void k_zero(float* p, int n) {
  int i = blockIdx.x * blockDim.x + threadIdx.x;
  if (i < n) p[i] = 0.f;
}

__global__ void k_deg(const int* ei, float* deg) {
  int e = blockIdx.x * blockDim.x + threadIdx.x;
  if (e < NEDGE) atomicAdd(&deg[ei[NEDGE + e]], 1.0f);
}

// h = x @ in_W + in_b   [8000,16]@[16,128]  (negligible FLOPs -> simple)
__global__ void k_in_proj(const float* __restrict__ x, const float* __restrict__ W,
                          const float* __restrict__ b, float* __restrict__ h) {
  int i = blockIdx.x * blockDim.x + threadIdx.x;
  if (i >= N_NODES * H) return;
  int n = i >> 7, j = i & 127;
  const float* xr = x + n * DIN;
  float s = b[j];
#pragma unroll
  for (int k = 0; k < DIN; ++k) s += xr[k] * W[k * H + j];
  h[i] = s;
}

// Pre-swizzle conv_W2[l] ([128,2048] fp32, index [K][n]) into bf16 WMMA
// B-fragment order: pack[((nt*4+kt)*32+lane)*16 + j] = B[kt*32 + (lane>>4)*16 + j][nt*16 + (lane&15)]
__global__ void k_pack_w2(const float* __restrict__ W2, __bf16* __restrict__ pack) {
  int idx = blockIdx.x * blockDim.x + threadIdx.x;
  if (idx >= H * MDIM * H) return;   // 262144
  int j    = idx & 15;
  int lane = (idx >> 4) & 31;
  int kt   = (idx >> 9) & 3;
  int nt   = idx >> 11;
  int K = kt * 32 + ((lane >> 4) << 4) + j;
  int n = (nt << 4) + (lane & 15);
  pack[idx] = (__bf16)W2[K * (MDIM * H) + n];
}

// ------------------------------------------------- fused edge-message kernel
// Per block: 64 edges.  We1 = relu(eattr@W1+b1) built into LDS in A-fragment
// order; GEMM [64,128]@[128,2048] via v_wmma_f32_16x16x32_bf16; each C tile
// (column tile nt <=> h-index nt) is folded on the fly into
//   msgs[e,m] += (C[e,nt*16+m] + b2[nt*16+m]) * h_src[e][nt]
// accumulated per-lane in VGPRs, reduced across waves via LDS float atomics,
// then scattered to agg[dst] with global fp32 atomics.
__global__ __launch_bounds__(256)
void k_edge_msg(const int* __restrict__ ei, const float* __restrict__ eattr,
                const float* __restrict__ W1, const float* __restrict__ b1,
                const float* __restrict__ b2, const __bf16* __restrict__ w2p,
                const float* __restrict__ h, float* __restrict__ agg) {
  __shared__ __align__(32) __bf16 Apack[4][4][32][16]; // [et][kt][lane][16] 16KB
  __shared__ __align__(16) float  hT[H][HSTR];         // h_src transposed    34KB
  __shared__ float smsg[ETILE][MDIM];                  // 4KB
  __shared__ int   ssrc[ETILE], sdst[ETILE];

  const int tid  = threadIdx.x;
  const int base = blockIdx.x * ETILE;

  if (tid < ETILE)              ssrc[tid]          = ei[base + tid];
  else if (tid < 2 * ETILE)     sdst[tid - ETILE]  = ei[NEDGE + base + (tid - ETILE)];
  for (int i = tid; i < ETILE * MDIM; i += 256) smsg[i >> 4][i & 15] = 0.f;
  __syncthreads();

  // gather source-node features, transposed: hT[hh][e]
  for (int idx = tid; idx < ETILE * H; idx += 256) {
    int e = idx >> 7, hh = idx & 127;
    hT[hh][e] = h[ssrc[e] * H + hh];
  }
  // edge MLP stage 1 (+relu) directly into WMMA A-fragment layout (contiguous k = 16*half + elem, same order as the w2 pack)
  for (int idx = tid; idx < ETILE * H; idx += 256) {
    int e = idx >> 7, k = idx & 127;
    const float* ar = eattr + (size_t)(base + e) * EDF;
    float v = b1[k];
#pragma unroll
    for (int c = 0; c < EDF; ++c) v += ar[c] * W1[c * H + k];
    v = fmaxf(v, 0.f);
    int kt = k >> 5, K = k & 31;
    int half = K >> 4;
    int vv   = (K & 15) >> 1;
    Apack[e >> 4][kt][half * 16 + (e & 15)][vv * 2 + (K & 1)] = (__bf16)v;
  }
  __syncthreads();

  const int wave  = tid >> 5, lane = tid & 31;
  const int colm  = lane & 15;            // m  (C column within tile)
  const int ebase = (lane >> 4) << 3;     // C rows: ebase..ebase+7

  v8f macc[4];
#pragma unroll
  for (int et = 0; et < 4; ++et)
#pragma unroll
    for (int i = 0; i < 8; ++i) macc[et][i] = 0.f;

  for (int t = 0; t < 16; ++t) {
    int nt = wave * 16 + t;               // column tile == h index, 0..127
    v8f acc[4];
#pragma unroll
    for (int et = 0; et < 4; ++et)
#pragma unroll
      for (int i = 0; i < 8; ++i) acc[et][i] = 0.f;

#pragma unroll
    for (int kt = 0; kt < 4; ++kt) {
      const __bf16* bp = w2p + (((size_t)(nt * 4 + kt) * 32 + lane) << 4);
      v16bf bfrag = *(const v16bf*)bp;
      __builtin_prefetch(bp + 2048, 0, 1);   // stream next nt's fragments
#pragma unroll
      for (int et = 0; et < 4; ++et) {
        v16bf afrag = *(const v16bf*)(&Apack[et][kt][lane][0]);
        acc[et] = __builtin_amdgcn_wmma_f32_16x16x32_bf16(
            false, afrag, false, bfrag, (short)0, acc[et], false, false);
      }
    }
    float bias = b2[nt * MDIM + colm];
#pragma unroll
    for (int et = 0; et < 4; ++et) {
      const float* hp = &hT[nt][et * 16 + ebase];
      float4 ha = *(const float4*)hp;
      float4 hb = *(const float4*)(hp + 4);
      float hv[8] = {ha.x, ha.y, ha.z, ha.w, hb.x, hb.y, hb.z, hb.w};
#pragma unroll
      for (int i = 0; i < 8; ++i)
        macc[et][i] += (acc[et][i] + bias) * hv[i];
    }
  }
  // cross-wave combine (ds_add_f32), targets are lane-unique within a wave
#pragma unroll
  for (int et = 0; et < 4; ++et)
#pragma unroll
    for (int i = 0; i < 8; ++i)
      atomicAdd(&smsg[et * 16 + ebase + i][colm], macc[et][i]);
  __syncthreads();

  for (int idx = tid; idx < ETILE * MDIM; idx += 256) {
    int e = idx >> 4, m = idx & 15;
    atomicAdd(&agg[sdst[e] * MDIM + m], smsg[e][m]);
  }
}

// x_msg = agg/deg + h@root_W + conv_bias ;  h += x_msg@msg_W + msg_b
__global__ __launch_bounds__(128)
void k_node_update(const float* __restrict__ agg, const float* __restrict__ deg,
                   const float* __restrict__ rootW, const float* __restrict__ cbias,
                   const float* __restrict__ msgW, const float* __restrict__ msgb,
                   float* __restrict__ h) {
  __shared__ float hl[H];
  __shared__ float xm[MDIM];
  int n = blockIdx.x, j = threadIdx.x;
  hl[j] = h[n * H + j];
  __syncthreads();
  if (j < MDIM) {
    float s = agg[n * MDIM + j] / fmaxf(deg[n], 1.0f) + cbias[j];
    for (int hh = 0; hh < H; ++hh) s += hl[hh] * rootW[hh * MDIM + j];
    xm[j] = s;
  }
  __syncthreads();
  float v = hl[j] + msgb[j];
#pragma unroll
  for (int m = 0; m < MDIM; ++m) v += xm[m] * msgW[m * H + j];
  h[n * H + j] = v;
}

__global__ __launch_bounds__(256)
void k_bn_stats(const float* __restrict__ h, float* __restrict__ mu, float* __restrict__ var) {
  __shared__ float s1[256], s2[256];
  int j = blockIdx.x, tid = threadIdx.x;
  float a = 0.f, b = 0.f;
  for (int n = tid; n < N_NODES; n += 256) {
    float v = h[n * H + j];
    a += v; b += v * v;
  }
  s1[tid] = a; s2[tid] = b;
  __syncthreads();
  for (int off = 128; off > 0; off >>= 1) {
    if (tid < off) { s1[tid] += s1[tid + off]; s2[tid] += s2[tid + off]; }
    __syncthreads();
  }
  if (tid == 0) {
    float m = s1[0] / (float)N_NODES;
    mu[j]  = m;
    var[j] = s2[0] / (float)N_NODES - m * m;
  }
}

__global__ void k_bn_apply(float* __restrict__ h, const float* __restrict__ mu,
                           const float* __restrict__ var, const float* __restrict__ gamma,
                           const float* __restrict__ beta) {
  int i = blockIdx.x * blockDim.x + threadIdx.x;
  if (i >= N_NODES * H) return;
  int j = i & 127;
  float v = gamma[j] * (h[i] - mu[j]) * rsqrtf(var[j] + BN_EPS) + beta[j];
  h[i] = fmaxf(v, 0.f);
}

__global__ void k_out_proj(const float* __restrict__ h, const float* __restrict__ W,
                           const float* __restrict__ b, float* __restrict__ out) {
  int i = blockIdx.x * blockDim.x + threadIdx.x;
  if (i >= N_NODES * DOUT) return;
  int n = i >> 2, o = i & 3;
  const float* hr = h + n * H;
  float s = b[o];
#pragma unroll 16
  for (int k = 0; k < H; ++k) s += hr[k] * W[k * DOUT + o];
  out[i] = s;
}

// ------------------------------------------------------------------- launch
extern "C" void kernel_launch(void* const* d_in, const int* in_sizes, int n_in,
                              void* d_out, int out_size, void* d_ws, size_t ws_size,
                              hipStream_t stream) {
  const float* x     = (const float*)d_in[0];
  const int*   ei    = (const int*)  d_in[1];
  const float* eattr = (const float*)d_in[2];
  const float* inW   = (const float*)d_in[3];
  const float* inb   = (const float*)d_in[4];
  const float* cW1   = (const float*)d_in[5];
  const float* cb1   = (const float*)d_in[6];
  const float* cW2   = (const float*)d_in[7];
  const float* cb2   = (const float*)d_in[8];
  const float* rootW = (const float*)d_in[9];
  const float* cbias = (const float*)d_in[10];
  const float* gamma = (const float*)d_in[11];
  const float* beta  = (const float*)d_in[12];
  const float* msgW  = (const float*)d_in[13];
  const float* msgb  = (const float*)d_in[14];
  const float* outW  = (const float*)d_in[15];
  const float* outb  = (const float*)d_in[16];

  char* ws = (char*)d_ws;
  const size_t off_deg = 0;                         // N floats
  const size_t off_h   = 32768;                     // N*H floats (4 MB)
  const size_t off_agg = off_h   + 4096000;         // N*M floats
  const size_t off_w2p = off_agg + 524288;          // H*M*H bf16 (512 KB)
  const size_t off_mu  = off_w2p + 524288;
  float*  deg = (float*) (ws + off_deg);
  float*  h   = (float*) (ws + off_h);
  float*  agg = (float*) (ws + off_agg);
  __bf16* w2p = (__bf16*)(ws + off_w2p);
  float*  mu  = (float*) (ws + off_mu);
  float*  var = mu + H;

  k_zero<<<(N_NODES + 255) / 256, 256, 0, stream>>>(deg, N_NODES);
  k_deg<<<(NEDGE + 255) / 256, 256, 0, stream>>>(ei, deg);
  k_in_proj<<<(N_NODES * H + 255) / 256, 256, 0, stream>>>(x, inW, inb, h);

  for (int l = 0; l < NLAYERS; ++l) {
    k_pack_w2<<<(H * MDIM * H + 255) / 256, 256, 0, stream>>>(cW2 + (size_t)l * H * MDIM * H, w2p);
    k_zero<<<(N_NODES * MDIM + 255) / 256, 256, 0, stream>>>(agg, N_NODES * MDIM);
    k_edge_msg<<<NEDGE / ETILE, 256, 0, stream>>>(ei, eattr,
                                                  cW1 + (size_t)l * EDF * H,
                                                  cb1 + (size_t)l * H,
                                                  cb2 + (size_t)l * MDIM * H,
                                                  w2p, h, agg);
    k_node_update<<<N_NODES, 128, 0, stream>>>(agg, deg,
                                               rootW + (size_t)l * H * MDIM,
                                               cbias + (size_t)l * MDIM,
                                               msgW + (size_t)l * MDIM * H,
                                               msgb + (size_t)l * H, h);
    k_bn_stats<<<H, 256, 0, stream>>>(h, mu, var);
    k_bn_apply<<<(N_NODES * H + 255) / 256, 256, 0, stream>>>(h, mu, var,
                                                              gamma + (size_t)l * H,
                                                              beta + (size_t)l * H);
  }
  k_out_proj<<<(N_NODES * DOUT + 255) / 256, 256, 0, stream>>>(h, outW, outb, (float*)d_out);
}
